// DengueGNN_33852932227575
// MI455X (gfx1250) — matrix operand order repaired and checked by reading, not compiled, not run on hardware
//
#include <hip/hip_runtime.h>
#include <hip/hip_bf16.h>
#include <math.h>

typedef __attribute__((ext_vector_type(16))) _Float16 v16h;
typedef __attribute__((ext_vector_type(8)))  float    v8f;

static constexpr int NN  = 50000;   // nodes
static constexpr int SEQ = 52;
static constexpr int FIN = 16;
static constexpr int HD  = 128;     // hidden
static constexpr float NEG_SLOPE = 0.2f;

// packed-weight geometry: 24 j-tiles (384/16) x 4 k-steps x 32 lanes x 16 halves
static constexpr int PW_FRAGS  = 24 * 4;          // fragments per matrix
static constexpr int PW_HALVES = PW_FRAGS * 32 * 16;  // 49152 halves per matrix

__device__ __forceinline__ float sigmoidf_(float x) { return 1.0f / (1.0f + __expf(-x)); }

__device__ __forceinline__ void atomicMaxFloat(float* addr, float val) {
    if (val >= 0.0f)
        atomicMax((int*)addr, __float_as_int(val));
    else
        atomicMin((unsigned int*)addr, __float_as_uint(val));
}

__device__ __forceinline__ v8f wmma16(v16h a, v16h b, v8f c) {
    // D = A(16x32 f16) * B(32x16 f16) + C(16x16 f32)
    return __builtin_amdgcn_wmma_f32_16x16x32_f16(false, a, false, b, (short)0, c, false, false);
}

// ---------------------------------------------------------------------------
// 0. pack W_ih / W_hh (f32 [384,128]) into f16 B-fragment lane order:
//    frag f = (jt*4 + kk); lane l: j = jt*16 + (l&15), k = kk*32 + (l>>4)*16 + i
//    dst[(f*32 + l)*16 + i]  ->  one contiguous 32B vector per lane
// ---------------------------------------------------------------------------
__global__ void pack_weights(const float* __restrict__ W_ih, const float* __restrict__ W_hh,
                             _Float16* __restrict__ pwi, _Float16* __restrict__ pwh)
{
    const int p = blockIdx.x * blockDim.x + threadIdx.x;   // (frag,lane) pairs, 2 matrices
    if (p >= 2 * PW_FRAGS * 32) return;
    const bool second = p >= PW_FRAGS * 32;
    const int  q    = second ? p - PW_FRAGS * 32 : p;
    const int  fid  = q >> 5;
    const int  lane = q & 31;
    const int  jt = fid >> 2, kk = fid & 3;
    const int  j  = jt * 16 + (lane & 15);
    const int  kb = kk * 32 + (lane >> 4) * 16;
    const float*   src = (second ? W_hh : W_ih) + (size_t)j * HD + kb;
    _Float16*      dst = (second ? pwh : pwi) + ((size_t)(fid * 32 + lane)) * 16;
#pragma unroll
    for (int i = 0; i < 16; ++i) dst[i] = (_Float16)src[i];
}

// ---------------------------------------------------------------------------
// 1. xp = x @ Wg  ;  a_s = xp @ att_src ; a_d = xp @ att_dst
//    one 128-thread block per node (K=16 is tiny; VALU is fine here)
// ---------------------------------------------------------------------------
__global__ void gat_linear(const float* __restrict__ x, const float* __restrict__ Wg,
                           const float* __restrict__ att_s, const float* __restrict__ att_d,
                           float* __restrict__ xp, float* __restrict__ a_s, float* __restrict__ a_d)
{
    __shared__ float xrow[FIN];
    __shared__ float red_s[4], red_d[4];
    const int n = blockIdx.x;
    const int c = threadIdx.x;          // 0..127
    if (c < FIN) xrow[c] = x[(size_t)n * FIN + c];
    __syncthreads();
    float acc = 0.0f;
#pragma unroll
    for (int k = 0; k < FIN; ++k) acc = fmaf(xrow[k], Wg[k * HD + c], acc);
    xp[(size_t)n * HD + c] = acc;

    float vs = acc * att_s[c];
    float vd = acc * att_d[c];
#pragma unroll
    for (int o = 16; o > 0; o >>= 1) { vs += __shfl_down(vs, o, 32); vd += __shfl_down(vd, o, 32); }
    const int lane = c & 31, w = c >> 5;
    if (lane == 0) { red_s[w] = vs; red_d[w] = vd; }
    __syncthreads();
    if (c == 0) {
        a_s[n] = red_s[0] + red_s[1] + red_s[2] + red_s[3];
        a_d[n] = red_d[0] + red_d[1] + red_d[2] + red_d[3];
    }
}

// ---------------------------------------------------------------------------
// 2. zero agg / denom, m = -inf
// ---------------------------------------------------------------------------
__global__ void attn_init(float* __restrict__ agg, float* __restrict__ mx, float* __restrict__ den)
{
    const int i = blockIdx.x * blockDim.x + threadIdx.x;
    if (i < NN * HD) agg[i] = 0.0f;
    if (i < NN) { mx[i] = -__builtin_inff(); den[i] = 0.0f; }
}

__global__ void zero_f32(float* __restrict__ p, int n)
{
    const int i = blockIdx.x * blockDim.x + threadIdx.x;
    if (i < n) p[i] = 0.0f;
}

__device__ __forceinline__ void edge_sd(const int* __restrict__ ei, int e, int E, int& s, int& d)
{
    if (e < E) { s = ei[e]; d = ei[E + e]; }
    else       { s = d = e - E; }            // self loops appended
}

// ---------------------------------------------------------------------------
// 3. segment max of leaky-relu edge scores
// ---------------------------------------------------------------------------
__global__ void edge_scores(const int* __restrict__ ei, const float* __restrict__ a_s,
                            const float* __restrict__ a_d, float* __restrict__ mx, int E, int E2)
{
    const int e = blockIdx.x * blockDim.x + threadIdx.x;
    if (e >= E2) return;
    int s, d; edge_sd(ei, e, E, s, d);
    const float sc = a_s[s] + a_d[d];
    const float lk = sc > 0.0f ? sc : NEG_SLOPE * sc;
    atomicMaxFloat(&mx[d], lk);
}

// ---------------------------------------------------------------------------
// 4. softmax denominator
// ---------------------------------------------------------------------------
__global__ void edge_sums(const int* __restrict__ ei, const float* __restrict__ a_s,
                          const float* __restrict__ a_d, const float* __restrict__ mx,
                          float* __restrict__ den, int E, int E2)
{
    const int e = blockIdx.x * blockDim.x + threadIdx.x;
    if (e >= E2) return;
    int s, d; edge_sd(ei, e, E, s, d);
    const float sc = a_s[s] + a_d[d];
    const float lk = sc > 0.0f ? sc : NEG_SLOPE * sc;
    atomicAdd(&den[d], __expf(lk - mx[d]));
}

// ---------------------------------------------------------------------------
// 5. weighted aggregation: agg[dst] += alpha * xp[src]
//    one wave32 per edge; 32 lanes x float4 == 128 columns
// ---------------------------------------------------------------------------
__global__ void edge_aggregate(const int* __restrict__ ei, const float* __restrict__ a_s,
                               const float* __restrict__ a_d, const float* __restrict__ mx,
                               const float* __restrict__ den, const float* __restrict__ xp,
                               float* __restrict__ agg, int E, int E2)
{
    const int gid  = blockIdx.x * blockDim.x + threadIdx.x;
    const int e    = gid >> 5;
    const int lane = gid & 31;
    if (e >= E2) return;
    int s, d; edge_sd(ei, e, E, s, d);
    const float sc = a_s[s] + a_d[d];
    const float lk = sc > 0.0f ? sc : NEG_SLOPE * sc;
    const float alpha = __expf(lk - mx[d]) / den[d];

    const float4 v = ((const float4*)(xp + (size_t)s * HD))[lane];
    float* dst = agg + (size_t)d * HD + lane * 4;
    atomicAdd(dst + 0, alpha * v.x);
    atomicAdd(dst + 1, alpha * v.y);
    atomicAdd(dst + 2, alpha * v.z);
    atomicAdd(dst + 3, alpha * v.w);
}

// ---------------------------------------------------------------------------
// 6. fused GRU cell with WMMA (pre-packed f16 weights):
//    spatial = relu(agg + bias_g)                           (f16 on the fly)
//    gi = spatial @ W_ih^T ; gh = h @ W_hh^T                (v_wmma f16->f32)
//    h_new = (1-z)*n + z*h
//    block = 256 thr (8 waves), 16-node row tile; wave w owns H-cols [16w,16w+16)
// ---------------------------------------------------------------------------
__global__ void gru_fused(const float* __restrict__ agg, const float* __restrict__ bias_g,
                          const float* __restrict__ h,
                          const _Float16* __restrict__ pwi, const _Float16* __restrict__ pwh,
                          const float* __restrict__ b_ih, const float* __restrict__ b_hh,
                          float* __restrict__ hnew)
{
    __shared__ _Float16 xs[16][HD + 8];   // padded to dodge bank conflicts
    __shared__ _Float16 hs[16][HD + 8];
    const int tid = threadIdx.x;
    const int nb  = blockIdx.x * 16;      // node row base (N divisible by 16)

    for (int i = tid; i < 16 * HD; i += blockDim.x) {
        const int r = i >> 7, c = i & 127;
        const float xv = agg[(size_t)(nb + r) * HD + c] + bias_g[c];
        xs[r][c] = (_Float16)(xv > 0.0f ? xv : 0.0f);
        hs[r][c] = (_Float16)h[(size_t)(nb + r) * HD + c];
    }
    __syncthreads();

    const int lane = tid & 31;
    const int cc   = tid >> 5;        // 0..7 -> H column tile
    const int nl   = lane & 15;       // row (A) / column (B,C) within tile
    const int hi   = lane >> 4;

    const v16h* BI = (const v16h*)pwi;
    const v16h* BH = (const v16h*)pwh;

    v8f ir = {}, iz = {}, in_ = {}, hr = {}, hz = {}, hn = {};

#pragma unroll
    for (int kk = 0; kk < 4; ++kk) {              // K = 128 = 4 x 32
        // A fragments (16x32 f16): lanes0-15 K{0-15}, lanes16-31 K{16-31}
        const int ka = kk * 32 + (hi ? 16 : 0);
        v16h ax, ah;
#pragma unroll
        for (int i = 0; i < 8; ++i) {
            ax[i]     = xs[nl][ka + i];
            ax[8 + i] = xs[nl][ka + 8 + i];
            ah[i]     = hs[nl][ka + i];
            ah[8 + i] = hs[nl][ka + 8 + i];
        }
        // B fragments: single 32B vector per lane from packed layout
        // frag id = ((g*8 + cc)*4 + kk); jt = g*8 + cc
#pragma unroll
        for (int g = 0; g < 3; ++g) {             // r, z, n gate blocks
            const int fidx = ((g * 8 + cc) * 4 + kk) * 32 + lane;
            const v16h bi = BI[fidx];
            const v16h bh = BH[fidx];
            if (g == 0)      { ir  = wmma16(ax, bi, ir);  hr = wmma16(ah, bh, hr); }
            else if (g == 1) { iz  = wmma16(ax, bi, iz);  hz = wmma16(ah, bh, hz); }
            else             { in_ = wmma16(ax, bi, in_); hn = wmma16(ah, bh, hn); }
        }
    }

    // elementwise GRU on the f32 accumulators
    // C layout: VGPR v -> M = v + 8*hi ; N = lane&15
    const int col = cc * 16 + nl;
    const float bir = b_ih[col],          bhr = b_hh[col];
    const float biz = b_ih[HD + col],     bhz = b_hh[HD + col];
    const float bin = b_ih[2 * HD + col], bhn = b_hh[2 * HD + col];
#pragma unroll
    for (int v = 0; v < 8; ++v) {
        const int row = nb + v + hi * 8;
        const float r  = sigmoidf_((ir[v] + bir) + (hr[v] + bhr));
        const float z  = sigmoidf_((iz[v] + biz) + (hz[v] + bhz));
        const float nv = tanhf((in_[v] + bin) + r * (hn[v] + bhn));
        const float ho = h[(size_t)row * HD + col];
        hnew[(size_t)row * HD + col] = (1.0f - z) * nv + z * ho;
    }
}

// ---------------------------------------------------------------------------
// 7. out = h @ W_fc^T + b_fc   (wave per node)
// ---------------------------------------------------------------------------
__global__ void fc_out(const float* __restrict__ h, const float* __restrict__ W_fc,
                       const float* __restrict__ b_fc, float* __restrict__ out)
{
    const int gid  = blockIdx.x * blockDim.x + threadIdx.x;
    const int node = gid >> 5;
    const int lane = gid & 31;
    if (node >= NN) return;
    float s = 0.0f;
#pragma unroll
    for (int i = 0; i < 4; ++i)
        s = fmaf(h[(size_t)node * HD + lane + 32 * i], W_fc[lane + 32 * i], s);
#pragma unroll
    for (int o = 16; o > 0; o >>= 1) s += __shfl_down(s, o, 32);
    if (lane == 0) out[node] = s + b_fc[0];
}

// ---------------------------------------------------------------------------
extern "C" void kernel_launch(void* const* d_in, const int* in_sizes, int n_in,
                              void* d_out, int out_size, void* d_ws, size_t ws_size,
                              hipStream_t stream)
{
    (void)n_in; (void)out_size; (void)ws_size;
    const float* x_seq   = (const float*)d_in[0];
    const int*   ei      = (const int*)  d_in[1];
    const float* Wg      = (const float*)d_in[2];
    const float* att_src = (const float*)d_in[3];
    const float* att_dst = (const float*)d_in[4];
    const float* bias_g  = (const float*)d_in[5];
    const float* W_ih    = (const float*)d_in[6];
    const float* W_hh    = (const float*)d_in[7];
    const float* b_ih    = (const float*)d_in[8];
    const float* b_hh    = (const float*)d_in[9];
    const float* W_fc    = (const float*)d_in[10];
    const float* b_fc    = (const float*)d_in[11];
    float*       out     = (float*)d_out;

    const int E  = in_sizes[1] / 2;
    const int E2 = E + NN;

    // workspace carve-up (floats, then f16 packed weights; 32B-aligned tail)
    float* ws  = (float*)d_ws;
    float* xp  = ws;                          // N*H
    float* a_s = xp  + (size_t)NN * HD;       // N
    float* a_d = a_s + NN;                    // N
    float* mx  = a_d + NN;                    // N
    float* den = mx  + NN;                    // N
    float* agg = den + NN;                    // N*H
    float* hA  = agg + (size_t)NN * HD;       // N*H
    float* hB  = hA  + (size_t)NN * HD;       // N*H
    _Float16* pwi = (_Float16*)(hB + (size_t)NN * HD);   // PW_HALVES
    _Float16* pwh = pwi + PW_HALVES;                     // PW_HALVES

    pack_weights<<<(2 * PW_FRAGS * 32 + 255) / 256, 256, 0, stream>>>(W_ih, W_hh, pwi, pwh);
    zero_f32<<<(NN * HD + 255) / 256, 256, 0, stream>>>(hA, NN * HD);

    float* hcur = hA;
    float* hnxt = hB;
    for (int t = 0; t < SEQ; ++t) {
        gat_linear<<<NN, 128, 0, stream>>>(x_seq + (size_t)t * NN * FIN, Wg,
                                           att_src, att_dst, xp, a_s, a_d);
        attn_init<<<(NN * HD + 255) / 256, 256, 0, stream>>>(agg, mx, den);
        edge_scores<<<(E2 + 255) / 256, 256, 0, stream>>>(ei, a_s, a_d, mx, E, E2);
        edge_sums<<<(E2 + 255) / 256, 256, 0, stream>>>(ei, a_s, a_d, mx, den, E, E2);
        edge_aggregate<<<(E2 + 7) / 8, 256, 0, stream>>>(ei, a_s, a_d, mx, den, xp, agg, E, E2);
        gru_fused<<<NN / 16, 256, 0, stream>>>(agg, bias_g, hcur, pwi, pwh, b_ih, b_hh, hnxt);
        float* tmp = hcur; hcur = hnxt; hnxt = tmp;
    }
    fc_out<<<(NN + 7) / 8, 256, 0, stream>>>(hcur, W_fc, b_fc, out);
}
